// GeneEnvAttentionModelWithoutEffectSeparation_15006615734198
// MI455X (gfx1250) — matrix operand order repaired and checked by reading, not compiled, not run on hardware
//
#include <hip/hip_runtime.h>
#include <math.h>

typedef __attribute__((ext_vector_type(16))) _Float16 v16h;
typedef __attribute__((ext_vector_type(8)))  _Float16 v8h;
typedef __attribute__((ext_vector_type(4)))  _Float16 v4h;
typedef __attribute__((ext_vector_type(8)))  float    v8f;

#define EPI_NONE    0
#define EPI_GELU    1
#define EPI_TANHMUL 2
#define EPI_ADD     3

__device__ __forceinline__ float gelu_erf(float x) {
    return 0.5f * x * (1.0f + erff(x * 0.70710678118654752440f));
}

// ---------------------------------------------------------------------------
// Tiled WMMA GEMM: C[M,N] = epi(A[M,K] @ W[K,N] + bias[N], other)
// Block = 128 threads = 4 waves. Block tile = 64(M) x 64(N), K step 32.
// Wave w owns N-strip [w*16, w*16+16); 4 accumulators cover M 0..63.
// LDS layouts are arranged so every WMMA fragment is two ds_load_b128:
//   As row-major   [m][k]  (row stride 40 f16 = 80 B, 16B-aligned runs)
//   Bs column-major[n][k]  (fragment = contiguous k-run at fixed n)
// ---------------------------------------------------------------------------
template<int EPI>
__global__ __launch_bounds__(128)
void gemm_wmma(const float* __restrict__ A, const float* __restrict__ W,
               const float* __restrict__ bias, const float* __restrict__ other,
               float* __restrict__ C, int M, int N, int K, int ldc)
{
    __shared__ _Float16 As[64][40];   // 64 rows x 32 k (+pad), 80B row stride
    __shared__ _Float16 Bs[64][40];   // 64 cols x 32 k (+pad), transposed tile

    const int tid  = threadIdx.x;
    const int wave = tid >> 5;
    const int lane = tid & 31;
    const int half = lane >> 4;       // 0: lanes 0-15, 1: lanes 16-31
    const int lm   = lane & 15;
    const int rowBase = blockIdx.y * 64;
    const int colBase = blockIdx.x * 64;

    v8f acc[4] = {};

    for (int k0 = 0; k0 < K; k0 += 32) {
        if (k0 + 32 < K) {   // prefetch next K-tile (global_prefetch_b8)
            __builtin_prefetch(&A[(rowBase + (tid >> 1)) * K + k0 + 32 + (tid & 1) * 16], 0, 1);
            __builtin_prefetch(&W[(k0 + 32 + (tid >> 2)) * N + colBase + (tid & 3) * 16], 0, 1);
        }
        // stage A tile: 64x32 floats, float4 per thread x4 -> packed f16 (ds_store_b64)
        #pragma unroll
        for (int i = 0; i < 4; i++) {
            int f = tid + i * 128;          // float4 index, 0..511
            int r = f >> 3;                 // 64 rows
            int c = (f & 7) * 4;            // 0..28
            const float4 av = *(const float4*)&A[(rowBase + r) * K + k0 + c];
            v4h h; h[0] = (_Float16)av.x; h[1] = (_Float16)av.y;
                   h[2] = (_Float16)av.z; h[3] = (_Float16)av.w;
            *(v4h*)&As[r][c] = h;
        }
        // stage B tile: 32x64 floats, float4 per thread x4 -> transposed f16
        #pragma unroll
        for (int i = 0; i < 4; i++) {
            int f = tid + i * 128;          // float4 index, 0..511
            int r = f >> 4;                 // 32 k-rows
            int c = (f & 15) * 4;           // 0..60
            const float4 bv = *(const float4*)&W[(k0 + r) * N + colBase + c];
            Bs[c + 0][r] = (_Float16)bv.x;
            Bs[c + 1][r] = (_Float16)bv.y;
            Bs[c + 2][r] = (_Float16)bv.z;
            Bs[c + 3][r] = (_Float16)bv.w;
        }
        __syncthreads();

        // B fragment: lane holds B[k = half*16 + e][n = wave*16 + lm]
        const v8h* bp = (const v8h*)&Bs[wave * 16 + lm][half * 16];
        v8h blo = bp[0], bhi = bp[1];
        v16h bfrag = __builtin_shufflevector(blo, bhi,
            0, 1, 2, 3, 4, 5, 6, 7, 8, 9, 10, 11, 12, 13, 14, 15);

        // A fragments: lane holds A[m][k = half*16 + e]
        #pragma unroll
        for (int mt = 0; mt < 4; mt++) {
            int row = mt * 16 + lm;
            v8h alo = *(const v8h*)&As[row][half * 16];
            v8h ahi = *(const v8h*)&As[row][half * 16 + 8];
            v16h afrag = __builtin_shufflevector(alo, ahi,
                0, 1, 2, 3, 4, 5, 6, 7, 8, 9, 10, 11, 12, 13, 14, 15);
            acc[mt] = __builtin_amdgcn_wmma_f32_16x16x32_f16(false, afrag, false, bfrag,
                                                             (short)0, acc[mt], false, false);
        }
        __syncthreads();
    }

    // epilogue: C/D layout -> VGPR v holds row (half*8 + v), col = lane&15
    const int col = colBase + wave * 16 + lm;
    const float bv = bias[col];
    #pragma unroll
    for (int mt = 0; mt < 4; mt++) {
        #pragma unroll
        for (int v = 0; v < 8; v++) {
            int r = rowBase + mt * 16 + half * 8 + v;
            float x = acc[mt][v] + bv;
            if (EPI == EPI_GELU) {
                x = gelu_erf(x);
            } else if (EPI == EPI_TANHMUL) {
                x = other[r * ldc + col] * tanhf(x);
            } else if (EPI == EPI_ADD) {
                x += other[r * ldc + col];
            }
            C[r * ldc + col] = x;
        }
    }
}

// ---------------------------------------------------------------------------
// Row LayerNorm (population variance, eps=1e-5), optional fused erf-GELU
// ---------------------------------------------------------------------------
template<int GELU>
__global__ __launch_bounds__(256)
void layernorm_kernel(const float* __restrict__ X, const float* __restrict__ g,
                      const float* __restrict__ b, float* __restrict__ Y, int D)
{
    const int row = blockIdx.x;
    const float* x = X + row * D;
    float* y = Y + row * D;
    __shared__ float s1[256], s2[256];
    float lsum = 0.f, lsq = 0.f;
    for (int i = threadIdx.x; i < D; i += 256) {
        float v = x[i]; lsum += v; lsq += v * v;
    }
    s1[threadIdx.x] = lsum; s2[threadIdx.x] = lsq;
    __syncthreads();
    for (int s = 128; s > 0; s >>= 1) {
        if (threadIdx.x < s) {
            s1[threadIdx.x] += s1[threadIdx.x + s];
            s2[threadIdx.x] += s2[threadIdx.x + s];
        }
        __syncthreads();
    }
    float mean = s1[0] / (float)D;
    float var  = s2[0] / (float)D - mean * mean;
    float rstd = rsqrtf(var + 1e-5f);
    for (int i = threadIdx.x; i < D; i += 256) {
        float v = (x[i] - mean) * rstd * g[i] + b[i];
        if (GELU) v = gelu_erf(v);
        y[i] = v;
    }
}

// ---------------------------------------------------------------------------
// Small helpers (gate GEMV, softmax/top-2, MoE combine, aux loss, pred head)
// ---------------------------------------------------------------------------
__global__ void gate_logits_kernel(const float* __restrict__ x, const float* __restrict__ gw,
                                   const float* __restrict__ gb, float* __restrict__ logits,
                                   int B, int K, int E)
{
    int t = blockIdx.x * blockDim.x + threadIdx.x;
    if (t >= B * E) return;
    int b = t / E, e = t % E;
    const float* xr = x + b * K;
    float s = gb[e];
    for (int k = 0; k < K; k++) s += xr[k] * gw[k * E + e];
    logits[t] = s;
}

__global__ void gating_kernel(const float* __restrict__ logits,
                              float* __restrict__ probs, float* __restrict__ comb, int B)
{
    int b = blockIdx.x * blockDim.x + threadIdx.x;
    if (b >= B) return;
    float l[8];
    float mx = -1e30f;
    for (int e = 0; e < 8; e++) { l[e] = logits[b * 8 + e]; mx = fmaxf(mx, l[e]); }
    float s = 0.f;
    for (int e = 0; e < 8; e++) { l[e] = __expf(l[e] - mx); s += l[e]; }
    float p[8];
    for (int e = 0; e < 8; e++) { p[e] = l[e] / s; probs[b * 8 + e] = p[e]; }
    int i0 = 0;
    for (int e = 1; e < 8; e++) if (p[e] > p[i0]) i0 = e;
    int i1 = (i0 == 0) ? 1 : 0;
    for (int e = 0; e < 8; e++) if (e != i0 && p[e] > p[i1]) i1 = e;
    float ws = p[i0] + p[i1];
    for (int e = 0; e < 8; e++) comb[b * 8 + e] = 0.f;
    comb[b * 8 + i0] = p[i0] / ws;
    comb[b * 8 + i1] = p[i1] / ws;
}

__global__ void zero_kernel(float* __restrict__ p, int n) {
    int t = blockIdx.x * blockDim.x + threadIdx.x;
    if (t < n) p[t] = 0.f;
}

__global__ void moe_accum_kernel(const float* __restrict__ y, const float* __restrict__ comb,
                                 float* __restrict__ acc, int B, int FD, int e)
{
    int t = blockIdx.x * blockDim.x + threadIdx.x;
    if (t >= B * FD) return;
    int b = t / FD;
    acc[t] += comb[b * 8 + e] * y[t];
}

__global__ void aux_loss_kernel(const float* __restrict__ probs, const float* __restrict__ comb,
                                float* __restrict__ out, int B)
{
    __shared__ float red[256];
    __shared__ float impv[8], loadv[8];
    for (int e = 0; e < 8; e++) {
        float s = 0.f;
        for (int r = threadIdx.x; r < B; r += 256) s += probs[r * 8 + e];
        red[threadIdx.x] = s; __syncthreads();
        for (int st = 128; st > 0; st >>= 1) {
            if (threadIdx.x < st) red[threadIdx.x] += red[threadIdx.x + st];
            __syncthreads();
        }
        if (threadIdx.x == 0) impv[e] = red[0];
        __syncthreads();
        s = 0.f;
        for (int r = threadIdx.x; r < B; r += 256) s += (comb[r * 8 + e] > 0.f) ? 1.f : 0.f;
        red[threadIdx.x] = s; __syncthreads();
        for (int st = 128; st > 0; st >>= 1) {
            if (threadIdx.x < st) red[threadIdx.x] += red[threadIdx.x + st];
            __syncthreads();
        }
        if (threadIdx.x == 0) loadv[e] = red[0] / (float)B;
        __syncthreads();
    }
    if (threadIdx.x == 0) {
        float mi = 0.f, ml = 0.f;
        for (int e = 0; e < 8; e++) { mi += impv[e]; ml += loadv[e]; }
        mi /= 8.f; ml /= 8.f;
        float vi = 0.f, vl = 0.f;
        for (int e = 0; e < 8; e++) {
            float d = impv[e] - mi; vi += d * d;
            d = loadv[e] - ml;      vl += d * d;
        }
        vi /= 7.f; vl /= 7.f;   // ddof=1
        float li = sqrtf(vi) / (mi + 1e-6f); li *= li;
        float ll = sqrtf(vl) / (ml + 1e-6f); ll *= ll;
        out[0] = li + ll;
    }
}

__global__ void pred_kernel(const float* __restrict__ h, const float* __restrict__ w,
                            const float* __restrict__ b, float* __restrict__ out, int B, int K)
{
    int r = blockIdx.x * blockDim.x + threadIdx.x;
    if (r >= B) return;
    const float* x = h + r * K;
    float s = b[0];
    for (int k = 0; k < K; k++) s += x[k] * w[k];
    out[r] = s;
}

// ---------------------------------------------------------------------------
// Host orchestration
// ---------------------------------------------------------------------------
static inline dim3 gemm_grid(int M, int N) { return dim3(N / 64, M / 64, 1); }

extern "C" void kernel_launch(void* const* d_in, const int* in_sizes, int n_in,
                              void* d_out, int out_size, void* d_ws, size_t ws_size,
                              hipStream_t stream)
{
    (void)in_sizes; (void)n_in; (void)out_size; (void)ws_size;
    const int B = 1024, S = 4096, E_ = 64, WU = 1024, EU = 256;
    const int SA = 512, EA = 128, FD = 512, MH = 1024, NE = 8, F1 = 512, F2 = 256;

    const float* snp    = (const float*)d_in[0];
    const float* env    = (const float*)d_in[1];
    const float* snp_w1 = (const float*)d_in[2];  const float* snp_b1 = (const float*)d_in[3];
    const float* snp_lng= (const float*)d_in[4];  const float* snp_lnb= (const float*)d_in[5];
    const float* snp_w2 = (const float*)d_in[6];  const float* snp_b2 = (const float*)d_in[7];
    const float* env_w1 = (const float*)d_in[8];  const float* env_b1 = (const float*)d_in[9];
    const float* env_lng= (const float*)d_in[10]; const float* env_lnb= (const float*)d_in[11];
    const float* env_w2 = (const float*)d_in[12]; const float* env_b2 = (const float*)d_in[13];
    const float* snp_pw = (const float*)d_in[14]; const float* snp_pb = (const float*)d_in[15];
    const float* env_pw = (const float*)d_in[16]; const float* env_pb = (const float*)d_in[17];
    const float* snp_wv = (const float*)d_in[18]; const float* snp_bv = (const float*)d_in[19];
    const float* snp_wo = (const float*)d_in[20]; const float* snp_bo = (const float*)d_in[21];
    const float* snp_ng = (const float*)d_in[22]; const float* snp_nb = (const float*)d_in[23];
    const float* env_wv = (const float*)d_in[24]; const float* env_bv = (const float*)d_in[25];
    const float* env_wo = (const float*)d_in[26]; const float* env_bo = (const float*)d_in[27];
    const float* env_ng = (const float*)d_in[28]; const float* env_nb = (const float*)d_in[29];
    const float* snp_fw = (const float*)d_in[30]; const float* snp_fb = (const float*)d_in[31];
    const float* env_fw = (const float*)d_in[32]; const float* env_fb = (const float*)d_in[33];
    const float* fus_ng = (const float*)d_in[34]; const float* fus_nb = (const float*)d_in[35];
    const float* fus_fw = (const float*)d_in[36]; const float* fus_fb = (const float*)d_in[37];
    const float* gate_w = (const float*)d_in[38]; const float* gate_b = (const float*)d_in[39];
    const float* ew1    = (const float*)d_in[40]; const float* eb1    = (const float*)d_in[41];
    const float* elg    = (const float*)d_in[42]; const float* elb    = (const float*)d_in[43];
    const float* ew2    = (const float*)d_in[44]; const float* eb2    = (const float*)d_in[45];
    const float* moe_ng = (const float*)d_in[46]; const float* moe_nb = (const float*)d_in[47];
    const float* fn1_w  = (const float*)d_in[48]; const float* fn1_b  = (const float*)d_in[49];
    const float* fn1_g  = (const float*)d_in[50]; const float* fn1_bb = (const float*)d_in[51];
    const float* fn2_w  = (const float*)d_in[52]; const float* fn2_b  = (const float*)d_in[53];
    const float* fn2_g  = (const float*)d_in[54]; const float* fn2_bb = (const float*)d_in[55];
    const float* pred_w = (const float*)d_in[56]; const float* pred_b = (const float*)d_in[57];

    float* out = (float*)d_out;

    // bump allocator over workspace
    float* ws = (float*)d_ws;
    size_t off = 0;
    auto alloc = [&](size_t n) { float* p = ws + off; off += n; return p; };

    float* th1     = alloc((size_t)B * WU);
    float* th2     = alloc((size_t)B * WU);
    float* snp_eff = alloc((size_t)B * S);
    float* eh1b    = alloc((size_t)B * EU);
    float* eh2b    = alloc((size_t)B * EU);
    float* env_eff = alloc((size_t)B * E_);
    float* sp      = alloc((size_t)B * SA);
    float* sv      = alloc((size_t)B * SA);
    float* sres    = alloc((size_t)B * SA);
    float* sf      = alloc((size_t)B * SA);
    float* ep      = alloc((size_t)B * EA);
    float* ev      = alloc((size_t)B * EA);
    float* eres    = alloc((size_t)B * EA);
    float* ef      = alloc((size_t)B * EA);
    float* fused   = alloc((size_t)B * 2 * FD);
    float* fusedn  = alloc((size_t)B * 2 * FD);
    float* ff      = alloc((size_t)B * FD);
    float* logits  = alloc((size_t)B * NE);
    float* probs   = alloc((size_t)B * NE);
    float* comb    = alloc((size_t)B * NE);
    float* xh1     = alloc((size_t)B * MH);
    float* xh2     = alloc((size_t)B * MH);
    float* xy      = alloc((size_t)B * FD);
    float* moe_acc = alloc((size_t)B * FD);
    float* moe_out = alloc((size_t)B * FD);
    float* f1      = alloc((size_t)B * F1);
    float* f1n     = alloc((size_t)B * F1);
    float* f2      = alloc((size_t)B * F2);
    float* f2n     = alloc((size_t)B * F2);

    const dim3 blk(128);
    const dim3 blk256(256);

    // ---- snp gating branch: tanh(gelu(LN(x@w1+b1))@w2+b2) * x
    gemm_wmma<EPI_NONE><<<gemm_grid(B, WU), blk, 0, stream>>>(snp, snp_w1, snp_b1, nullptr, th1, B, WU, S, WU);
    layernorm_kernel<1><<<B, blk256, 0, stream>>>(th1, snp_lng, snp_lnb, th2, WU);
    gemm_wmma<EPI_TANHMUL><<<gemm_grid(B, S), blk, 0, stream>>>(th2, snp_w2, snp_b2, snp, snp_eff, B, S, WU, S);

    // ---- env gating branch
    gemm_wmma<EPI_NONE><<<gemm_grid(B, EU), blk, 0, stream>>>(env, env_w1, env_b1, nullptr, eh1b, B, EU, E_, EU);
    layernorm_kernel<1><<<B, blk256, 0, stream>>>(eh1b, env_lng, env_lnb, eh2b, EU);
    gemm_wmma<EPI_TANHMUL><<<gemm_grid(B, E_), blk, 0, stream>>>(eh2b, env_w2, env_b2, env, env_eff, B, E_, EU, E_);

    // ---- projections
    gemm_wmma<EPI_NONE><<<gemm_grid(B, SA), blk, 0, stream>>>(snp_eff, snp_pw, snp_pb, nullptr, sp, B, SA, S, SA);
    gemm_wmma<EPI_NONE><<<gemm_grid(B, EA), blk, 0, stream>>>(env_eff, env_pw, env_pb, nullptr, ep, B, EA, E_, EA);

    // ---- seq_len==1 attention: LN(x + out_proj(v_proj(x)))
    gemm_wmma<EPI_NONE><<<gemm_grid(B, SA), blk, 0, stream>>>(sp, snp_wv, snp_bv, nullptr, sv, B, SA, SA, SA);
    gemm_wmma<EPI_ADD><<<gemm_grid(B, SA), blk, 0, stream>>>(sv, snp_wo, snp_bo, sp, sres, B, SA, SA, SA);
    layernorm_kernel<0><<<B, blk256, 0, stream>>>(sres, snp_ng, snp_nb, sf, SA);
    gemm_wmma<EPI_NONE><<<gemm_grid(B, EA), blk, 0, stream>>>(ep, env_wv, env_bv, nullptr, ev, B, EA, EA, EA);
    gemm_wmma<EPI_ADD><<<gemm_grid(B, EA), blk, 0, stream>>>(ev, env_wo, env_bo, ep, eres, B, EA, EA, EA);
    layernorm_kernel<0><<<B, blk256, 0, stream>>>(eres, env_ng, env_nb, ef, EA);

    // ---- fusion: concat(sf@snp_fw+fb, ef@env_fw+fb) -> LN -> gelu(@fus_fw+fb)
    gemm_wmma<EPI_NONE><<<gemm_grid(B, FD), blk, 0, stream>>>(sf, snp_fw, snp_fb, nullptr, fused, B, FD, SA, 2 * FD);
    gemm_wmma<EPI_NONE><<<gemm_grid(B, FD), blk, 0, stream>>>(ef, env_fw, env_fb, nullptr, fused + FD, B, FD, EA, 2 * FD);
    layernorm_kernel<0><<<B, blk256, 0, stream>>>(fused, fus_ng, fus_nb, fusedn, 2 * FD);
    gemm_wmma<EPI_GELU><<<gemm_grid(B, FD), blk, 0, stream>>>(fusedn, fus_fw, fus_fb, nullptr, ff, B, FD, 2 * FD, FD);

    // ---- MoE gating
    gate_logits_kernel<<<(B * NE + 255) / 256, blk256, 0, stream>>>(ff, gate_w, gate_b, logits, B, FD, NE);
    gating_kernel<<<(B + 255) / 256, blk256, 0, stream>>>(logits, probs, comb, B);
    aux_loss_kernel<<<1, blk256, 0, stream>>>(probs, comb, out + B, B);

    // ---- experts (dense grouped compute, weighted accumulate)
    zero_kernel<<<(B * FD + 255) / 256, blk256, 0, stream>>>(moe_acc, B * FD);
    for (int e = 0; e < NE; e++) {
        const float* w1 = ew1 + (size_t)e * FD * MH;
        const float* b1 = eb1 + (size_t)e * MH;
        const float* lg = elg + (size_t)e * MH;
        const float* lb = elb + (size_t)e * MH;
        const float* w2 = ew2 + (size_t)e * MH * FD;
        const float* b2 = eb2 + (size_t)e * FD;
        gemm_wmma<EPI_NONE><<<gemm_grid(B, MH), blk, 0, stream>>>(ff, w1, b1, nullptr, xh1, B, MH, FD, MH);
        layernorm_kernel<1><<<B, blk256, 0, stream>>>(xh1, lg, lb, xh2, MH);
        gemm_wmma<EPI_NONE><<<gemm_grid(B, FD), blk, 0, stream>>>(xh2, w2, b2, nullptr, xy, B, FD, MH, FD);
        moe_accum_kernel<<<(B * FD + 255) / 256, blk256, 0, stream>>>(xy, comb, moe_acc, B, FD, e);
    }
    layernorm_kernel<0><<<B, blk256, 0, stream>>>(moe_acc, moe_ng, moe_nb, moe_out, FD);

    // ---- feature network + head
    gemm_wmma<EPI_NONE><<<gemm_grid(B, F1), blk, 0, stream>>>(moe_out, fn1_w, fn1_b, nullptr, f1, B, F1, FD, F1);
    layernorm_kernel<1><<<B, blk256, 0, stream>>>(f1, fn1_g, fn1_bb, f1n, F1);
    gemm_wmma<EPI_NONE><<<gemm_grid(B, F2), blk, 0, stream>>>(f1n, fn2_w, fn2_b, nullptr, f2, B, F2, F1, F2);
    layernorm_kernel<1><<<B, blk256, 0, stream>>>(f2, fn2_g, fn2_bb, f2n, F2);
    pred_kernel<<<(B + 255) / 256, blk256, 0, stream>>>(f2n, pred_w, pred_b, out, B, F2);
}
